// SparseUNet4_85813446574842
// MI455X (gfx1250) — matrix operand order repaired and checked by reading, not compiled, not run on hardware
//
#include <hip/hip_runtime.h>

typedef __attribute__((ext_vector_type(16))) _Float16 v16h;
typedef __attribute__((ext_vector_type(8)))  _Float16 v8h;
typedef __attribute__((ext_vector_type(8)))  float    v8f;
typedef __attribute__((ext_vector_type(4)))  unsigned int v4u;
typedef __attribute__((ext_vector_type(8)))  int      v8i;
typedef __attribute__((ext_vector_type(4)))  int      v4i;

#define BN_EPS 1e-5f

// ---------------- f32 -> f16 convert ----------------
__global__ void k_cvt_f32_f16(const float* __restrict__ src, _Float16* __restrict__ dst, int n) {
  int i = blockIdx.x * 256 + threadIdx.x;
  if (i < n) dst[i] = (_Float16)src[i];
}

// ---------------- weight packing into WMMA B-layout ----------------
// Packed layout: wp[(((nt*ksteps)+ks)*32 + lane)*16 + (2v+h)]
//   B 32x16 f16 layout: lanes 0-15 hold K {0..7,16..23}, lanes 16-31 hold K {8..15,24..31},
//   element e holds K = 8*(lane>=16) + (e&7) + 16*(e>=8), N = lane & 15.
__global__ void k_pack_w(const float* __restrict__ w, _Float16* __restrict__ wp,
                         int Ktot, int Cout, int ksteps, int total) {
  int idx = blockIdx.x * 256 + threadIdx.x;
  if (idx >= total) return;
  int e    = idx & 15;
  int lane = (idx >> 4) & 31;
  int rest = idx >> 9;
  int ks   = rest % ksteps;
  int nt   = rest / ksteps;
  int v = e >> 1, h = e & 1;
  int kk = ks * 32 + ((e & 8) ? 16 : 0) + ((lane & 16) ? 8 : 0) + (e & 7); (void)v; (void)h;
  int n  = nt * 16 + (lane & 15);
  float val = (kk < Ktot) ? w[(size_t)kk * Cout + n] : 0.0f;
  wp[idx] = (_Float16)val;
}

// ---------------- TDM: stage one B panel (rows x 1024B, row stride in halves) into LDS ----------------
__device__ __forceinline__ void tdm_load_b(const _Float16* gsrc, unsigned lds_off,
                                           int rows, int rowStrideHalves) {
  unsigned long long ga = (unsigned long long)(size_t)gsrc;
  v4u g0;
  g0.x = 1u;                                                    // count=1, user descriptor
  g0.y = lds_off;                                               // LDS byte address
  g0.z = (unsigned)(ga & 0xffffffffu);                          // global_addr[31:0]
  g0.w = ((unsigned)(ga >> 32) & 0x01ffffffu) | 0x80000000u;    // global_addr[56:32] | type=2
  v8i g1;
  g1[0] = (int)(1u << 16);                    // data_size=1 (2 bytes), no multicast
  g1[1] = (int)(512u << 16);                  // tensor_dim0[15:0] = 512 elements
  g1[2] = (int)((unsigned)rows << 16);        // tensor_dim0 hi=0 | tensor_dim1 = rows
  g1[3] = (int)(512u << 16);                  // tensor_dim1 hi=0 | tile_dim0 = 512
  g1[4] = rows;                               // tile_dim1 = rows | tile_dim2 = 0
  g1[5] = rowStrideHalves;                    // tensor_dim0_stride[31:0]
  g1[6] = 0;                                  // stride0 hi | tensor_dim1_stride lo
  g1[7] = 0;
  v4i z4 = {0, 0, 0, 0};
#if defined(__clang_major__) && (__clang_major__ >= 23)
  v8i z8 = {};
  __builtin_amdgcn_tensor_load_to_lds(g0, g1, z4, z4, z8, 0);
#else
  __builtin_amdgcn_tensor_load_to_lds(g0, g1, z4, z4, 0);
#endif
}

// ---------------- implicit-GEMM conv via WMMA ----------------
// normal conv: i = o*stride + k - 1          (kernel 3, pad 1)
// transposed (stride-2 'SAME', zero-insertion): t = o + k - 1, valid iff t even, i = t/2
__device__ __forceinline__ bool src_coord(int o, int k, int stride, int tr, int Din, int& i) {
  if (tr) {
    int t = o + k - 1;
    if (t < 0 || (t & 1)) return false;
    i = t >> 1;
    return i < Din;
  }
  i = o * stride + k - 1;
  return (unsigned)i < (unsigned)Din;
}

// Load 8 consecutive K-halves (one kernel tap, 8 consecutive channels) as one
// aligned 16-byte vector, or zeros when out of bounds (Cin % 8 == 0 path).
__device__ __forceinline__ v8h gather8(const _Float16* __restrict__ act,
                                       int kb, int Ktot, int Cin, int cinLog2,
                                       int od, int oh, int ow,
                                       int Din, int stride, int tr) {
  v8h z = {};
  if (kb >= Ktot) return z;
  const int koff = kb >> cinLog2;
  const int ci   = kb & (Cin - 1);
  const int kd = koff / 9;
  const int r1 = koff - kd * 9;
  const int kh = r1 / 3;
  const int kw = r1 - kh * 3;
  int id, ih, iw;
  if (src_coord(od, kd, stride, tr, Din, id) &&
      src_coord(oh, kh, stride, tr, Din, ih) &&
      src_coord(ow, kw, stride, tr, Din, iw)) {
    return *(const v8h*)(act + (((size_t)id * Din + ih) * Din + iw) * Cin + ci);
  }
  return z;
}

__global__ __launch_bounds__(128)
void k_conv_wmma(const _Float16* __restrict__ act, const _Float16* __restrict__ wp,
                 float* __restrict__ out,
                 int Din, int Dout, int Cin, int cinLog2, int Cout,
                 int M, int ksteps, int stride, int tr, int ntsub, int useTdm) {
  __shared__ _Float16 lds_b[2][2048];          // double-buffered B panel (2 x 4KB)
  const int lane  = threadIdx.x;
  const int wave  = threadIdx.y;
  const int mtile = blockIdx.x * 4 + wave;
  const int mbase = mtile * 16;
  const int ntb   = blockIdx.y * ntsub;
  const int Ktot  = 27 * Cin;

  // A-matrix row for this lane (16-bit A 16x32 layout: M = lane&15,
  // lanes 0-15 carry K {0..7,16..23}, lanes 16-31 carry K {8..15,24..31})
  const int  mrow   = mbase + (lane & 15);
  const bool mvalid = (mrow < M);
  const int  mc = mvalid ? mrow : 0;
  const int  od = mc / (Dout * Dout);
  const int  r0 = mc - od * Dout * Dout;
  const int  oh = r0 / Dout;
  const int  ow = r0 - oh * Dout;
  const int  khalf8 = (lane & 16) ? 8 : 0;

  v8f acc[4] = {};

  if (useTdm) {
    // ===== Cin >= 32: tap-outer loop + TDM-staged, double-buffered B in LDS =====
    const int spt = Cin >> 5;                   // K-steps per kernel tap
    const unsigned ldsOff0 = (unsigned)(size_t)&lds_b[0][0];
    const unsigned ldsOff1 = (unsigned)(size_t)&lds_b[1][0];
    const _Float16* wpb = wp + (size_t)ntb * ksteps * 512;

    if (wave == 0)
      tdm_load_b(wpb, ldsOff0, ntsub, ksteps * 512);

    int ks = 0;
    for (int tap = 0; tap < 27; ++tap) {
      const int kd = tap / 9;
      const int r1 = tap - kd * 9;
      const int kh = r1 / 3;
      const int kw = r1 - kh * 3;
      int id, ih, iw;
      const bool ok = mvalid &&
                      src_coord(od, kd, stride, tr, Din, id) &&
                      src_coord(oh, kh, stride, tr, Din, ih) &&
                      src_coord(ow, kw, stride, tr, Din, iw);
      const _Float16* pbase =
          ok ? act + (((size_t)id * Din + ih) * Din + iw) * Cin : (const _Float16*)0;

      for (int s = 0; s < spt; ++s, ++ks) {
        __syncthreads();                        // previous buffer reads complete
        if (wave == 0) {
          if (ks + 1 < ksteps) {
            tdm_load_b(wpb + (size_t)(ks + 1) * 512,
                       ((ks + 1) & 1) ? ldsOff1 : ldsOff0, ntsub, ksteps * 512);
            __builtin_amdgcn_s_wait_tensorcnt(1);   // step ks has landed
          } else {
            __builtin_amdgcn_s_wait_tensorcnt(0);
          }
        }
        __syncthreads();                        // B panel for ks visible to all waves

        v8h g0v = {}, g1v = {};
        if (ok) {
          g0v = *(const v8h*)(pbase + s * 32 + khalf8);
          g1v = *(const v8h*)(pbase + s * 32 + 16 + khalf8);
        }
        v16h a = __builtin_shufflevector(g0v, g1v, 0, 1, 2, 3, 4, 5, 6, 7,
                                                   8, 9, 10, 11, 12, 13, 14, 15);
        const _Float16* lb = &lds_b[ks & 1][0];
#pragma unroll
        for (int t = 0; t < 4; ++t) {
          if (t < ntsub) {
            v16h b = *(const v16h*)(lb + t * 512 + lane * 16);   // ds_load_b128 x2
            acc[t] = __builtin_amdgcn_wmma_f32_16x16x32_f16(
                false, a, false, b, (short)0, acc[t], false, false);
          }
        }
      }
    }
  } else {
    // ===== Cin == 1 or 16: generic path, B straight from global =====
    if (mbase >= M) return;                     // wave-uniform exit (no barriers here)
    for (int ks = 0; ks < ksteps; ++ks) {
      v16h a;
      if (Cin >= 8) {
        v8h g0 = {}, g1 = {};
        if (mvalid) {
          g0 = gather8(act, ks * 32 + khalf8,      Ktot, Cin, cinLog2, od, oh, ow, Din, stride, tr);
          g1 = gather8(act, ks * 32 + 16 + khalf8, Ktot, Cin, cinLog2, od, oh, ow, Din, stride, tr);
        }
        a = __builtin_shufflevector(g0, g1, 0, 1, 2, 3, 4, 5, 6, 7,
                                            8, 9, 10, 11, 12, 13, 14, 15);
      } else {
#pragma unroll
        for (int v = 0; v < 8; ++v) {
#pragma unroll
          for (int h = 0; h < 2; ++h) {
            const int kk = ks * 32 + ((v & 4) ? 16 : 0) + khalf8 + 2 * (v & 3) + h;
            _Float16 val = (_Float16)0.0f;
            if (mvalid && kk < Ktot) {
              const int koff = kk >> cinLog2;
              const int ci   = kk & (Cin - 1);
              const int kd = koff / 9;
              const int r1 = koff - kd * 9;
              const int kh = r1 / 3;
              const int kw = r1 - kh * 3;
              int id, ih, iw;
              if (src_coord(od, kd, stride, tr, Din, id) &&
                  src_coord(oh, kh, stride, tr, Din, ih) &&
                  src_coord(ow, kw, stride, tr, Din, iw)) {
                val = act[(((size_t)id * Din + ih) * Din + iw) * Cin + ci];
              }
            }
            a[2 * v + h] = val;
          }
        }
      }
#pragma unroll
      for (int t = 0; t < 4; ++t) {
        if (t < ntsub) {
          const _Float16* bsrc = wp + ((size_t)((ntb + t) * ksteps + ks) * 32 + lane) * 16;
          v16h b = *(const v16h*)bsrc;
          acc[t] = __builtin_amdgcn_wmma_f32_16x16x32_f16(
              false, a, false, b, (short)0, acc[t], false, false);
        }
      }
      if (ks + 1 < ksteps)
        __builtin_prefetch(wp + ((size_t)(ntb * ksteps + ks + 1) * 32 + lane) * 16, 0, 3);
    }
  }

  // D layout: VGPR r -> M = r + 8*(lane>=16), N = lane&15
#pragma unroll
  for (int t = 0; t < 4; ++t) {
    if (t < ntsub) {
#pragma unroll
      for (int rr = 0; rr < 8; ++rr) {
        const int mg = mbase + rr + ((lane & 16) ? 8 : 0);
        if (mg < M) out[(size_t)mg * Cout + (ntb + t) * 16 + (lane & 15)] = acc[t][rr];
      }
    }
  }
}

// ---------------- BatchNorm statistics (per channel over all voxels) ----------------
__global__ __launch_bounds__(256)
void k_bn_stats(const float* __restrict__ x, float* __restrict__ stats, int M, int C) {
  const int c = blockIdx.x;
  float s = 0.f, ss = 0.f;
  for (int i = threadIdx.x; i < M; i += 256) {
    float v = x[(size_t)i * C + c];
    s += v; ss += v * v;
  }
  __shared__ float sh1[256];
  __shared__ float sh2[256];
  sh1[threadIdx.x] = s; sh2[threadIdx.x] = ss;
  __syncthreads();
  for (int o = 128; o > 0; o >>= 1) {
    if ((int)threadIdx.x < o) {
      sh1[threadIdx.x] += sh1[threadIdx.x + o];
      sh2[threadIdx.x] += sh2[threadIdx.x + o];
    }
    __syncthreads();
  }
  if (threadIdx.x == 0) {
    float inv = 1.0f / (float)M;
    float mu  = sh1[0] * inv;
    float var = sh2[0] * inv - mu * mu;
    stats[2 * c]     = mu;
    stats[2 * c + 1] = rsqrtf(var + BN_EPS);
  }
}

// ---------------- fused BN + ReLU, writing f16 (optionally into concat slot) and/or f32 ----------------
__global__ void k_bn_apply(const float* __restrict__ x, const float* __restrict__ stats,
                           const float* __restrict__ g, const float* __restrict__ b,
                           _Float16* dst16, int dstStride, int dstOff,
                           float* dstf, int M, int C) {
  long long i = (long long)blockIdx.x * 256 + threadIdx.x;
  long long total = (long long)M * C;
  if (i >= total) return;
  int c = (int)(i % C);
  long long m = i / C;
  float mu = stats[2 * c], is = stats[2 * c + 1];
  float y = (x[i] - mu) * is * g[c] + b[c];
  y = fmaxf(y, 0.0f);
  if (dst16) dst16[m * dstStride + dstOff + c] = (_Float16)y;
  if (dstf)  dstf[i] = y;
}

// ---------------- skip-connection channel copy into concat buffer ----------------
__global__ void k_copy_ch(const _Float16* __restrict__ src, _Float16* __restrict__ dst,
                          int M, int C, int dstStride, int dstOff) {
  long long i = (long long)blockIdx.x * 256 + threadIdx.x;
  long long total = (long long)M * C;
  if (i >= total) return;
  int c = (int)(i % C);
  long long m = i / C;
  dst[m * dstStride + dstOff + c] = src[m * C + c];
}

// =====================================================================
extern "C" void kernel_launch(void* const* d_in, const int* in_sizes, int n_in,
                              void* d_out, int out_size, void* d_ws, size_t ws_size,
                              hipStream_t stream) {
  (void)in_sizes; (void)n_in; (void)out_size; (void)ws_size;

  const int S1 = 96, S2 = 48, S3 = 24, S4 = 12, S5 = 6;
  const int M1 = S1*S1*S1, M2 = S2*S2*S2, M3 = S3*S3*S3, M4 = S4*S4*S4, M5 = S5*S5*S5;

  char* ws = (char*)d_ws;
  size_t off = 0;
  auto alloc = [&](size_t bytes) -> char* {
    char* p = ws + off;
    off += (bytes + 255) & ~(size_t)255;
    return p;
  };

  // f16 activation buffers
  _Float16* xh   = (_Float16*)alloc((size_t)M1 * 1   * 2);
  _Float16* e1   = (_Float16*)alloc((size_t)M1 * 16  * 2);
  _Float16* t2   = (_Float16*)alloc((size_t)M2 * 32  * 2);
  _Float16* e2   = (_Float16*)alloc((size_t)M2 * 32  * 2);
  _Float16* t3   = (_Float16*)alloc((size_t)M3 * 64  * 2);
  _Float16* e3   = (_Float16*)alloc((size_t)M3 * 64  * 2);
  _Float16* t4   = (_Float16*)alloc((size_t)M4 * 128 * 2);
  _Float16* e4   = (_Float16*)alloc((size_t)M4 * 128 * 2);
  _Float16* t5   = (_Float16*)alloc((size_t)M5 * 256 * 2);
  _Float16* bt   = (_Float16*)alloc((size_t)M5 * 256 * 2);
  _Float16* cat4 = (_Float16*)alloc((size_t)M4 * 256 * 2);
  _Float16* d4   = (_Float16*)alloc((size_t)M4 * 128 * 2);
  _Float16* cat3 = (_Float16*)alloc((size_t)M3 * 128 * 2);
  _Float16* d3   = (_Float16*)alloc((size_t)M3 * 64  * 2);
  _Float16* cat2 = (_Float16*)alloc((size_t)M2 * 64  * 2);
  _Float16* d2   = (_Float16*)alloc((size_t)M2 * 32  * 2);
  _Float16* cat1 = (_Float16*)alloc((size_t)M1 * 32  * 2);
  // f32 conv scratch (max M1*16) + stats
  float* convf = (float*)alloc((size_t)M1 * 16 * 4);
  float* stats = (float*)alloc(512 * 4);

  struct PW { _Float16* p; int ksteps; int Cin; int Cout; };
  auto packW = [&](int inIdx, int Cin, int Cout) -> PW {
    int Ktot = 27 * Cin;
    int ksteps = (Ktot + 31) / 32;
    int ntiles = Cout / 16;
    int total  = ntiles * ksteps * 512;
    _Float16* p = (_Float16*)alloc((size_t)total * 2);
    k_pack_w<<<(total + 255) / 256, 256, 0, stream>>>(
        (const float*)d_in[inIdx], p, Ktot, Cout, ksteps, total);
    return PW{p, ksteps, Cin, Cout};
  };

  // pack all 17 conv weights into WMMA B layout
  PW enc1w = packW(1, 1, 16);
  PW e2d = packW(4, 16, 32),   e2c = packW(7, 32, 32);
  PW e3d = packW(10, 32, 64),  e3c = packW(13, 64, 64);
  PW e4d = packW(16, 64, 128), e4c = packW(19, 128, 128);
  PW btd = packW(22, 128, 256), btc = packW(25, 256, 256);
  PW u4 = packW(28, 256, 128), u3 = packW(31, 128, 64);
  PW u2 = packW(34, 64, 32),   u1 = packW(37, 32, 16);
  PW dc4 = packW(40, 256, 128), dc3 = packW(43, 128, 64);
  PW dc2 = packW(46, 64, 32),   dc1 = packW(49, 32, 16);

  auto conv = [&](const _Float16* act, const PW& w, int Din, int Dout, int stride, int tr,
                  int gIdx, int bIdx,
                  _Float16* dst16, int dstStride, int dstOff, float* dstf) {
    int M = Dout * Dout * Dout;
    int cinLog2 = 0; while ((1 << cinLog2) < w.Cin) cinLog2++;
    int ntiles = w.Cout / 16;
    int ntsub  = (ntiles >= 4) ? 4 : ntiles;     // 16->1, 32->2, >=64->4
    int mtiles = (M + 15) / 16;
    int useTdm = (w.Cin >= 32) ? 1 : 0;
    dim3 grid((mtiles + 3) / 4, ntiles / ntsub);
    k_conv_wmma<<<grid, dim3(32, 4), 0, stream>>>(
        act, w.p, convf, Din, Dout, w.Cin, cinLog2, w.Cout, M, w.ksteps, stride, tr,
        ntsub, useTdm);
    k_bn_stats<<<w.Cout, 256, 0, stream>>>(convf, stats, M, w.Cout);
    long long total = (long long)M * w.Cout;
    k_bn_apply<<<(int)((total + 255) / 256), 256, 0, stream>>>(
        convf, stats, (const float*)d_in[gIdx], (const float*)d_in[bIdx],
        dst16, dstStride, dstOff, dstf, M, w.Cout);
  };
  auto copyCh = [&](const _Float16* src, _Float16* dst, int M, int C, int dstStride, int dstOff) {
    long long total = (long long)M * C;
    k_copy_ch<<<(int)((total + 255) / 256), 256, 0, stream>>>(src, dst, M, C, dstStride, dstOff);
  };

  // ---- input to f16 ----
  k_cvt_f32_f16<<<(M1 + 255) / 256, 256, 0, stream>>>((const float*)d_in[0], xh, M1);

  // ---- encoder ----
  conv(xh, enc1w, S1, S1, 1, 0, 2, 3,  e1, 16, 0, nullptr);     // e1: 96^3 x16
  conv(e1, e2d,   S1, S2, 2, 0, 5, 6,  t2, 32, 0, nullptr);
  conv(t2, e2c,   S2, S2, 1, 0, 8, 9,  e2, 32, 0, nullptr);     // e2: 48^3 x32
  conv(e2, e3d,   S2, S3, 2, 0, 11, 12, t3, 64, 0, nullptr);
  conv(t3, e3c,   S3, S3, 1, 0, 14, 15, e3, 64, 0, nullptr);    // e3: 24^3 x64
  conv(e3, e4d,   S3, S4, 2, 0, 17, 18, t4, 128, 0, nullptr);
  conv(t4, e4c,   S4, S4, 1, 0, 20, 21, e4, 128, 0, nullptr);   // e4: 12^3 x128
  conv(e4, btd,   S4, S5, 2, 0, 23, 24, t5, 256, 0, nullptr);
  conv(t5, btc,   S5, S5, 1, 0, 26, 27, bt, 256, 0, nullptr);   // bt: 6^3 x256

  // ---- decoder (upconv writes into concat[:,0:C], skip copied to concat[:,C:2C]) ----
  conv(bt, u4, S5, S4, 2, 1, 29, 30, cat4, 256, 0, nullptr);
  copyCh(e4, cat4, M4, 128, 256, 128);
  conv(cat4, dc4, S4, S4, 1, 0, 41, 42, d4, 128, 0, nullptr);

  conv(d4, u3, S4, S3, 2, 1, 32, 33, cat3, 128, 0, nullptr);
  copyCh(e3, cat3, M3, 64, 128, 64);
  conv(cat3, dc3, S3, S3, 1, 0, 44, 45, d3, 64, 0, nullptr);

  conv(d3, u2, S3, S2, 2, 1, 35, 36, cat2, 64, 0, nullptr);
  copyCh(e2, cat2, M2, 32, 64, 32);
  conv(cat2, dc2, S2, S2, 1, 0, 47, 48, d2, 32, 0, nullptr);

  conv(d2, u1, S2, S1, 2, 1, 38, 39, cat1, 32, 0, nullptr);
  copyCh(e1, cat1, M1, 16, 32, 16);
  conv(cat1, dc1, S1, S1, 1, 0, 50, 51, nullptr, 0, 0, (float*)d_out);  // final d1 -> f32 out
}
